// GATBackbone_63522566308389
// MI455X (gfx1250) — matrix operand order repaired and checked by reading, not compiled, not run on hardware
//
#include <hip/hip_runtime.h>
#include <hip/hip_bf16.h>

// ---------------------------------------------------------------------------
// GAT backbone for MI455X (gfx1250, wave32, WMMA).
//   * a_dst via X @ (W_dst @ att_dst)        -> no full W_dst GEMM
//   * edge projection factored through aggregation:
//       sum_e w*(edge_attr@W_edge) = (scatter_e w*edge_attr) @ W_edge
//   * dense GEMMs: v_wmma_f32_16x16x32_bf16, fp32 accumulation
//   * WMMA fragments fed by 16-byte ds_load_b128; tile staging uses
//     unconditional vector loads with row-clamped addresses (OOB rows
//     only feed OOB outputs, which are never stored).
// Assumptions honored by all call sites: K % 32 == 0, Ncols % 64 == 0.
// ---------------------------------------------------------------------------

#define NN     20000
#define EE     160000
#define HEADS  4

typedef __attribute__((ext_vector_type(16))) __bf16          v16bf;
typedef __attribute__((ext_vector_type(16))) unsigned short  v16u;
typedef __attribute__((ext_vector_type(8)))  unsigned short  v8u;
typedef __attribute__((ext_vector_type(8)))  float           v8f;
typedef __attribute__((ext_vector_type(4)))  float           v4f;

static __device__ __forceinline__ unsigned short f32_to_bf16(float f) {
    unsigned u = __float_as_uint(f);
    u += 0x7FFFu + ((u >> 16) & 1u);   // round to nearest even
    return (unsigned short)(u >> 16);
}

static __device__ __forceinline__ v16bf as_v16bf(v16u u) {
    union { v16u u; v16bf b; } c; c.u = u; return c.b;
}

static __device__ __forceinline__ v16u cat8(v8u lo, v8u hi) {
    return __builtin_shufflevector(lo, hi, 0,1,2,3,4,5,6,7,8,9,10,11,12,13,14,15);
}

static __device__ __forceinline__ void atomicMaxFloat(float* addr, float v) {
    if (v >= 0.0f) atomicMax((int*)addr, __float_as_int(v));
    else           atomicMin((unsigned int*)addr, __float_as_uint(v));
}

// ---------------------------------------------------------------------------
// Tiled bf16-WMMA GEMM:  C = A(MxK) * B(KxN)  [+= if accumulate]
// Block tile 128x64, 8 waves, each wave computes a 16x64 strip (4 subtiles).
// ---------------------------------------------------------------------------
#define TILE_M 128
#define TILE_N 64
#define TILE_K 32
#define LDS_STRIDE 40   // ushorts; 80 B row pitch, multiple of 16 B

__global__ __launch_bounds__(256)
void gemm_bf16_wmma(const float* __restrict__ A, int lda,
                    const float* __restrict__ B, int ldb,
                    float* __restrict__ C, int ldc,
                    int M, int Ncols, int K, int accumulate)
{
    // A tile: row-major  As[row][k]   (fragment = 2 contiguous 8-elem runs)
    // B tile: col-major  BsT[col][k]  (fragment = 2 contiguous 8-elem runs)
    __shared__ __align__(16) unsigned short As [TILE_M][LDS_STRIDE];
    __shared__ __align__(16) unsigned short BsT[TILE_N][LDS_STRIDE];

    const int tid  = threadIdx.x;
    const int lane = tid & 31;
    const int wave = tid >> 5;
    const int m0   = blockIdx.x * TILE_M;
    const int n0   = blockIdx.y * TILE_N;

    const int col_lo = lane & 15;
    const int rbase  = (lane & 16) ? 8 : 0;

    // clamped staging row for this thread (unconditional loads; OOB rows
    // only ever feed OOB output rows, which are not stored)
    const int a_row  = tid >> 1;
    const int a_kb   = (tid & 1) * 16;
    int a_grow = m0 + a_row; if (a_grow > M - 1) a_grow = M - 1;
    const float* a_base = A + (size_t)a_grow * lda + a_kb;

    const int b_krow = tid >> 3;
    const int b_cb   = (tid & 7) * 8;
    const float* b_base = B + (size_t)b_krow * ldb + n0 + b_cb;

    v8f acc[4];
    if (accumulate) {
#pragma unroll
        for (int j = 0; j < 4; ++j) {
            const int col = n0 + j * 16 + col_lo;
#pragma unroll
            for (int r = 0; r < 8; ++r) {
                int row = m0 + wave * 16 + rbase + r;
                if (row > M - 1) row = M - 1;         // clamp: OOB rows unused
                acc[j][r] = C[(size_t)row * ldc + col];
            }
        }
    } else {
#pragma unroll
        for (int j = 0; j < 4; ++j)
#pragma unroll
            for (int r = 0; r < 8; ++r) acc[j][r] = 0.0f;
    }

    for (int k0 = 0; k0 < K; k0 += TILE_K) {
        // ---- stage A tile (128x32): 16 consecutive floats per thread,
        //      4x global_load_b128, two 16-byte ds_store_b128 ----
        {
            const v4f* sp = (const v4f*)(a_base + k0);
            const v4f f0 = sp[0], f1 = sp[1], f2 = sp[2], f3 = sp[3];
            v8u p0, p1;
#pragma unroll
            for (int i = 0; i < 4; ++i) {
                p0[i]     = f32_to_bf16(f0[i]);
                p0[4 + i] = f32_to_bf16(f1[i]);
                p1[i]     = f32_to_bf16(f2[i]);
                p1[4 + i] = f32_to_bf16(f3[i]);
            }
            *(v8u*)&As[a_row][a_kb]     = p0;
            *(v8u*)&As[a_row][a_kb + 8] = p1;
            if (k0 + TILE_K < K)
                __builtin_prefetch(a_base + k0 + TILE_K, 0, 1);  // global_prefetch_b8
        }
        // ---- stage B tile (32x64) transposed into LDS: coalesced global
        //      read along N (2x b128), strided 2-byte LDS stores ----
        {
            const v4f* sp = (const v4f*)(b_base + (size_t)k0 * ldb);
            const v4f g0 = sp[0], g1 = sp[1];
#pragma unroll
            for (int i = 0; i < 4; ++i) {
                BsT[b_cb + i][b_krow]     = f32_to_bf16(g0[i]);
                BsT[b_cb + 4 + i][b_krow] = f32_to_bf16(g1[i]);
            }
        }
        __syncthreads();

        // ---- A fragment: two ds_load_b128 ----
        const int arow  = wave * 16 + (lane & 15);
        const int khalf = (lane & 16) ? 8 : 0;
        const v8u alo = *(const v8u*)&As[arow][khalf];
        const v8u ahi = *(const v8u*)&As[arow][16 + khalf];
        const v16bf afrag = as_v16bf(cat8(alo, ahi));

        // ---- load all 4 B fragments first (8 ds_load_b128), then issue
        //      4 WMMAs back-to-back so dscnt waits can overlap ----
        const int kb2 = (lane & 16) ? 8 : 0;
        v16bf bfrag[4];
#pragma unroll
        for (int j = 0; j < 4; ++j) {
            const int bcol = j * 16 + (lane & 15);
            const v8u blo = *(const v8u*)&BsT[bcol][kb2];
            const v8u bhi = *(const v8u*)&BsT[bcol][16 + kb2];
            bfrag[j] = as_v16bf(cat8(blo, bhi));
        }
#pragma unroll
        for (int j = 0; j < 4; ++j) {
            acc[j] = __builtin_amdgcn_wmma_f32_16x16x32_bf16(
                         false, afrag, false, bfrag[j], (short)0, acc[j], false, false);
        }
        __syncthreads();
    }

    // ---- store (CDNA5 f32 C/D layout: col = lane&15, rows striped) ----
#pragma unroll
    for (int j = 0; j < 4; ++j) {
        const int col = n0 + j * 16 + col_lo;
#pragma unroll
        for (int r = 0; r < 8; ++r) {
            const int row = m0 + wave * 16 + rbase + r;
            if (row < M) C[(size_t)row * ldc + col] = acc[j][r];
        }
    }
}

// ---------------------------------------------------------------------------
// Per-layer helper kernels
// ---------------------------------------------------------------------------

// zero OUT / T / den, fill m with -inf
__global__ void init_kernel(float* out, long n_out, float* T, long n_T,
                            float* den, long n_den, float* m, long n_m)
{
    long i = (long)blockIdx.x * blockDim.x + threadIdx.x;
    long stride = (long)gridDim.x * blockDim.x;
    for (long j = i; j < n_out; j += stride) out[j] = 0.0f;
    for (long j = i; j < n_T;   j += stride) T[j]   = 0.0f;
    for (long j = i; j < n_den; j += stride) den[j] = 0.0f;
    for (long j = i; j < n_m;   j += stride) m[j]   = -3.0e38f;
}

// u_dst[d*H+h] = sum_c W_dst[d, h*C+c]*att_dst[h,c]
// v_edge[k*H+h] = sum_c W_edge[k, h*C+c]*att_edge[h,c]
__global__ void attvec_kernel(const float* __restrict__ Wd, const float* __restrict__ ad,
                              const float* __restrict__ We, const float* __restrict__ ae,
                              float* __restrict__ u_dst, float* __restrict__ v_edge,
                              int din, int C, int hc)
{
    int tid = blockIdx.x * blockDim.x + threadIdx.x;
    int nu = din * HEADS;
    if (tid < nu) {
        int d = tid >> 2, h = tid & 3;
        float s = 0.0f;
        for (int c = 0; c < C; ++c) s += Wd[(size_t)d * hc + h * C + c] * ad[h * C + c];
        u_dst[tid] = s;
    } else if (tid < nu + 64 * HEADS) {
        int t = tid - nu; int k = t >> 2, h = t & 3;
        float s = 0.0f;
        for (int c = 0; c < C; ++c) s += We[(size_t)k * hc + h * C + c] * ae[h * C + c];
        v_edge[t] = s;
    }
}

// a_src[n,h] = sum_c XS[n,h,c]*att_src[h,c];  a_dst[n,h] = sum_d X[n,d]*u_dst[d,h]
__global__ void node_att_kernel(const float* __restrict__ XS, const float* __restrict__ X,
                                const float* __restrict__ att_src, const float* __restrict__ u_dst,
                                float* __restrict__ a_src, float* __restrict__ a_dst,
                                int din, int C, int hc)
{
    int tid = blockIdx.x * blockDim.x + threadIdx.x;
    if (tid >= NN * HEADS) return;
    int n = tid >> 2, h = tid & 3;
    float s = 0.0f;
    for (int c = 0; c < C; ++c) s += XS[(size_t)n * hc + h * C + c] * att_src[h * C + c];
    a_src[tid] = s;
    float d = 0.0f;
    for (int k = 0; k < din; ++k) d += X[(size_t)n * din + k] * u_dst[k * HEADS + h];
    a_dst[tid] = d;
}

// alpha = leaky(a_dst[dst] + a_src[src] + edge_attr . v_edge); atomic max into m[dst]
__global__ void edge_alpha_kernel(const float* __restrict__ EA,
                                  const int* __restrict__ src, const int* __restrict__ dst,
                                  const float* __restrict__ a_src, const float* __restrict__ a_dst,
                                  const float* __restrict__ v_edge,
                                  float* __restrict__ alpha, float* __restrict__ mbuf)
{
    int tid = blockIdx.x * blockDim.x + threadIdx.x;
    if (tid >= EE * HEADS) return;
    int e = tid >> 2, h = tid & 3;
    float s = 0.0f;
    for (int k = 0; k < 64; ++k) s += EA[(size_t)e * 64 + k] * v_edge[k * HEADS + h];
    int dn = dst[e], sn = src[e];
    float a = a_dst[dn * HEADS + h] + a_src[sn * HEADS + h] + s;
    a = (a >= 0.0f) ? a : 0.2f * a;
    alpha[tid] = a;
    atomicMaxFloat(&mbuf[dn * HEADS + h], a);
}

// ex = exp(alpha - m[dst]);  den[dst] += ex
__global__ void edge_exp_kernel(float* __restrict__ alpha, const float* __restrict__ mbuf,
                                const int* __restrict__ dst, float* __restrict__ den)
{
    int tid = blockIdx.x * blockDim.x + threadIdx.x;
    if (tid >= EE * HEADS) return;
    int e = tid >> 2, h = tid & 3;
    int dn = dst[e];
    float ex = __expf(alpha[tid] - mbuf[dn * HEADS + h]);
    alpha[tid] = ex;
    atomicAdd(&den[dn * HEADS + h], ex);
}

// one workgroup per edge:
//   T[dst,h,k]   += w[h]*edge_attr[e,k]        (factored edge projection)
//   OUT[dst,h,c] += w[h]*XS[src,h,c]           (node message)
__global__ __launch_bounds__(256)
void edge_scatter_kernel(const float* __restrict__ ex, const float* __restrict__ den,
                         const int* __restrict__ src, const int* __restrict__ dst,
                         const float* __restrict__ EA, const float* __restrict__ XS,
                         float* __restrict__ T, float* __restrict__ OUT,
                         int C, int hc)
{
    const int e = blockIdx.x;
    __shared__ float ws[HEADS];
    __shared__ int sn_s, dn_s;
    if (threadIdx.x < HEADS) {
        int dn = dst[e];
        ws[threadIdx.x] = ex[e * HEADS + threadIdx.x] /
                          (den[dn * HEADS + threadIdx.x] + 1e-16f);
    }
    if (threadIdx.x == 0) { sn_s = src[e]; dn_s = dst[e]; }
    __syncthreads();

    const int t = threadIdx.x;
    {   // 4 heads * 64 edge features = 256 values
        const int h = t >> 6, k = t & 63;
        atomicAdd(&T[(size_t)dn_s * (HEADS * 64) + t], ws[h] * EA[(size_t)e * 64 + k]);
    }
    for (int j = t; j < hc; j += 256) {
        const int h = j / C;
        atomicAdd(&OUT[(size_t)dn_s * hc + j], ws[h] * XS[(size_t)sn_s * hc + j]);
    }
}

// bias + layernorm (+ optional leaky relu), one workgroup per node
__global__ __launch_bounds__(256)
void finalize_kernel(const float* __restrict__ OUT, const float* __restrict__ bias,
                     const float* __restrict__ g, const float* __restrict__ b,
                     float* __restrict__ Y, int hc, int apply_leaky)
{
    const int n = blockIdx.x;
    const int tid = threadIdx.x;
    __shared__ float s1[256], s2[256];
    float sum = 0.0f, sq = 0.0f;
    for (int j = tid; j < hc; j += 256) {
        float v = OUT[(size_t)n * hc + j] + bias[j];
        sum += v; sq += v * v;
    }
    s1[tid] = sum; s2[tid] = sq;
    __syncthreads();
    for (int s = 128; s > 0; s >>= 1) {
        if (tid < s) { s1[tid] += s1[tid + s]; s2[tid] += s2[tid + s]; }
        __syncthreads();
    }
    const float mu   = s1[0] / hc;
    const float var  = s2[0] / hc - mu * mu;
    const float rstd = rsqrtf(var + 1e-5f);
    for (int j = tid; j < hc; j += 256) {
        float v = OUT[(size_t)n * hc + j] + bias[j];
        v = (v - mu) * rstd * g[j] + b[j];
        if (apply_leaky) v = (v >= 0.0f) ? v : 0.2f * v;
        Y[(size_t)n * hc + j] = v;
    }
}

// ---------------------------------------------------------------------------
// Host orchestration
// ---------------------------------------------------------------------------
extern "C" void kernel_launch(void* const* d_in, const int* in_sizes, int n_in,
                              void* d_out, int out_size, void* d_ws, size_t ws_size,
                              hipStream_t stream)
{
    const float* x     = (const float*)d_in[0];
    const int*   ei    = (const int*)d_in[1];
    const float* eattr = (const float*)d_in[2];
    const int* src = ei;
    const int* dst = ei + EE;

    // workspace carve-out
    size_t off = 0;
    auto alloc = [&](size_t bytes) -> float* {
        float* p = (float*)((char*)d_ws + off);
        off += (bytes + 255) & ~(size_t)255;
        return p;
    };
    float* XS     = alloc((size_t)NN * 1024 * 4);
    float* OUT    = alloc((size_t)NN * 1024 * 4);
    float* XB0    = alloc((size_t)NN * 1024 * 4);
    float* XB1    = alloc((size_t)NN * 1024 * 4);
    float* T      = alloc((size_t)NN * HEADS * 64 * 4);
    float* a_src  = alloc((size_t)NN * HEADS * 4);
    float* a_dst  = alloc((size_t)NN * HEADS * 4);
    float* mbuf   = alloc((size_t)NN * HEADS * 4);
    float* den    = alloc((size_t)NN * HEADS * 4);
    float* alpha  = alloc((size_t)EE * HEADS * 4);
    float* u_dst  = alloc((size_t)1024 * HEADS * 4);
    float* v_edge = alloc((size_t)64 * HEADS * 4);

    const int dins[3] = {128, 1024, 1024};
    const int Cs[3]   = {256, 256, 64};

    for (int l = 0; l < 3; ++l) {
        const int din = dins[l];
        const int C   = Cs[l];
        const int hc  = HEADS * C;

        const float* W_src = (const float*)d_in[3 + 9 * l + 0];
        const float* W_dst = (const float*)d_in[3 + 9 * l + 1];
        const float* W_edg = (const float*)d_in[3 + 9 * l + 2];
        const float* at_s  = (const float*)d_in[3 + 9 * l + 3];
        const float* at_d  = (const float*)d_in[3 + 9 * l + 4];
        const float* at_e  = (const float*)d_in[3 + 9 * l + 5];
        const float* bias  = (const float*)d_in[3 + 9 * l + 6];
        const float* ln_g  = (const float*)d_in[3 + 9 * l + 7];
        const float* ln_b  = (const float*)d_in[3 + 9 * l + 8];

        const float* Xin  = (l == 0) ? x : ((l == 1) ? XB0 : XB1);
        float*       Xout = (l == 0) ? XB0 : ((l == 1) ? XB1 : (float*)d_out);

        // 1) reset accumulators
        init_kernel<<<4096, 256, 0, stream>>>(
            OUT, (long)NN * hc, T, (long)NN * HEADS * 64,
            den, (long)NN * HEADS, mbuf, (long)NN * HEADS);

        // 2) attention projection vectors (folds W_dst / W_edge with att vecs)
        {
            int tt = din * HEADS + 64 * HEADS;
            attvec_kernel<<<(tt + 255) / 256, 256, 0, stream>>>(
                W_dst, at_d, W_edg, at_e, u_dst, v_edge, din, C, hc);
        }

        // 3) XS = Xin @ W_src  (bf16 WMMA, fp32 acc)
        {
            dim3 grid((NN + TILE_M - 1) / TILE_M, hc / TILE_N);
            gemm_bf16_wmma<<<grid, 256, 0, stream>>>(
                Xin, din, W_src, hc, XS, hc, NN, hc, din, 0);
        }

        // 4) per-node attention scalars
        node_att_kernel<<<(NN * HEADS + 255) / 256, 256, 0, stream>>>(
            XS, Xin, at_s, u_dst, a_src, a_dst, din, C, hc);

        // 5) edge logits + segment max
        edge_alpha_kernel<<<(EE * HEADS + 255) / 256, 256, 0, stream>>>(
            eattr, src, dst, a_src, a_dst, v_edge, alpha, mbuf);

        // 6) exp + segment sum
        edge_exp_kernel<<<(EE * HEADS + 255) / 256, 256, 0, stream>>>(
            alpha, mbuf, dst, den);

        // 7) weighted scatter of messages + raw edge features
        edge_scatter_kernel<<<EE, 256, 0, stream>>>(
            alpha, den, src, dst, eattr, XS, T, OUT, C, hc);

        // 8) OUT[:, h*C:(h+1)*C] += T[:, h*64:(h+1)*64] @ W_edge[:, h*C:(h+1)*C]
        for (int h = 0; h < HEADS; ++h) {
            dim3 grid((NN + TILE_M - 1) / TILE_M, C / TILE_N);
            gemm_bf16_wmma<<<grid, 256, 0, stream>>>(
                T + h * 64, HEADS * 64, W_edg + h * C, hc, OUT + h * C, hc,
                NN, C, 64, 1);
        }

        // 9) bias + layernorm (+ leaky for layers 0,1)
        finalize_kernel<<<NN, 256, 0, stream>>>(
            OUT, bias, ln_g, ln_b, Xout, hc, (l < 2) ? 1 : 0);
    }
    (void)in_sizes; (void)n_in; (void)out_size; (void)ws_size;
}
